// MambaBlock_90460601189074
// MI455X (gfx1250) — matrix operand order repaired and checked by reading, not compiled, not run on hardware
//
#include <hip/hip_runtime.h>
#include <hip/hip_bf16.h>
#include <cstdint>
#include <cstddef>

// ---------- model constants ----------
#define D_MODEL   256
#define D_STATE   16
#define D_CONVK   4
#define D_INNER   512
#define DT_RANK   16
#define MLP_H     1024
#define BATCH     8
#define SEQL      2048
#define MROWS     (BATCH * SEQL)      // 16384

typedef _Float16 h16v __attribute__((ext_vector_type(16)));
typedef _Float16 h8v  __attribute__((ext_vector_type(8)));
typedef float    f8v  __attribute__((ext_vector_type(8)));

// ---------------------------------------------------------------------------
// Weight convert + transpose: W (N x K) f32  ->  Wt (K x N) f16
// ---------------------------------------------------------------------------
__global__ void k_convert_transpose(const float* __restrict__ W,
                                    _Float16* __restrict__ Wt,
                                    int N, int K) {
  int idx = blockIdx.x * blockDim.x + threadIdx.x;
  if (idx >= N * K) return;
  int n = idx / K;
  int k = idx - n * K;
  Wt[(size_t)k * N + n] = (_Float16)W[idx];
}

// ---------------------------------------------------------------------------
// RMSNorm over 256-wide rows -> f16 output. One wave32 per row, 8 elems/lane.
// ---------------------------------------------------------------------------
__global__ void k_rmsnorm_f16(const float* __restrict__ x,
                              const float* __restrict__ w,
                              _Float16* __restrict__ out,
                              int rows) {
  int gtid = blockIdx.x * blockDim.x + threadIdx.x;
  int wave = gtid >> 5;
  int lane = threadIdx.x & 31;
  if (wave >= rows) return;
  const float* xr = x + (size_t)wave * D_MODEL;
  float4 v0 = ((const float4*)xr)[lane * 2 + 0];
  float4 v1 = ((const float4*)xr)[lane * 2 + 1];
  float ss = v0.x * v0.x + v0.y * v0.y + v0.z * v0.z + v0.w * v0.w
           + v1.x * v1.x + v1.y * v1.y + v1.z * v1.z + v1.w * v1.w;
#pragma unroll
  for (int o = 16; o >= 1; o >>= 1) ss += __shfl_xor(ss, o, 32);
  float r = rsqrtf(ss * (1.0f / D_MODEL) + 1e-5f);
  int base = lane * 8;
  _Float16* orow = out + (size_t)wave * D_MODEL + base;
  const float* wr = w + base;
  float vals[8] = {v0.x, v0.y, v0.z, v0.w, v1.x, v1.y, v1.z, v1.w};
#pragma unroll
  for (int j = 0; j < 8; j++) orow[j] = (_Float16)(vals[j] * r * wr[j]);
}

// ---------------------------------------------------------------------------
// WMMA GEMM: C(MxN) = A(MxK,f16 row-major) * Bt(KxN,f16 row-major)
// One wave computes a (16*MSUB) x (16*NSUB) tile; MSUB/NSUB are compile-time
// so the inner loop has no guards (EXEC stays all-1 around every WMMA).
// __launch_bounds__(256, 2): relax VGPR cap so the unroll-2 double-buffered
// pipeline (acc 64 + A/B fragments ~ 190 VGPRs) lives without scratch spills.
// Optional bias[n], residual (MxN f32), f32 and/or f16 outputs.
// ---------------------------------------------------------------------------
template <int MSUB, int NSUB>
__global__ __launch_bounds__(256, 2)
void k_wmma_gemm(const _Float16* __restrict__ A,
                 const _Float16* __restrict__ Bt,
                 const float* __restrict__ bias,
                 const float* __restrict__ resid,
                 float* __restrict__ Cf,
                 _Float16* __restrict__ Ch,
                 int M, int N, int K) {
  int gtid = blockIdx.x * blockDim.x + threadIdx.x;
  int wave = gtid >> 5;
  int lane = threadIdx.x & 31;
  int tiles_m = M / (16 * MSUB);
  int tiles_n = (N + 16 * NSUB - 1) / (16 * NSUB);
  if (wave >= tiles_m * tiles_n) return;
  int tm = wave % tiles_m;
  int tn = wave / tiles_m;
  int m0 = tm * (16 * MSUB);
  int n0 = tn * (16 * NSUB);
  int hi = lane >> 4;          // half-wave select
  int lr = lane & 15;

  f8v acc[MSUB][NSUB];
#pragma unroll
  for (int ms = 0; ms < MSUB; ms++)
#pragma unroll
    for (int ns = 0; ns < NSUB; ns++) acc[ms][ns] = (f8v){};

  // A fragment (ISA 16-bit 16x32 layout): lane<16 holds K {0..7,16..23},
  // lane>=16 holds K {8..15,24..31} of its row -> two 16B loads per frag.
  const _Float16* arow[MSUB];
#pragma unroll
  for (int ms = 0; ms < MSUB; ms++)
    arow[ms] = A + (size_t)(m0 + ms * 16 + lr) * K + 8 * hi;

  // B fragment: lane holds column N=n0+lr of Bt, K {0..7,16..23} (lane<16) or {8..15,24..31}.
  const _Float16* bbase = Bt + n0 + lr;

#pragma unroll 2
  for (int k0 = 0; k0 < K; k0 += 32) {
    h16v b[NSUB];
#pragma unroll
    for (int ns = 0; ns < NSUB; ns++)
#pragma unroll
      for (int i = 0; i < 8; i++) {
        b[ns][i]     = bbase[(size_t)(k0 + 8 * hi + i) * N + ns * 16];
        b[ns][i + 8] = bbase[(size_t)(k0 + 16 + 8 * hi + i) * N + ns * 16];
      }

    h16v a[MSUB];
#pragma unroll
    for (int ms = 0; ms < MSUB; ms++) {
      h8v a0 = *(const h8v*)(arow[ms] + k0);
      h8v a1 = *(const h8v*)(arow[ms] + k0 + 16);
#pragma unroll
      for (int i = 0; i < 8; i++) { a[ms][i] = a0[i]; a[ms][i + 8] = a1[i]; }
    }

#pragma unroll
    for (int ms = 0; ms < MSUB; ms++)
#pragma unroll
      for (int ns = 0; ns < NSUB; ns++)
        acc[ms][ns] = __builtin_amdgcn_wmma_f32_16x16x32_f16(
            false, a[ms], false, b[ns], (short)0, acc[ms][ns], false, false);
  }

  // epilogue: C/D layout -> VGPR v holds (M = mbase + v + 8*hi, N = cb + lr)
#pragma unroll
  for (int ms = 0; ms < MSUB; ms++) {
#pragma unroll
    for (int ns = 0; ns < NSUB; ns++) {
      int n = n0 + ns * 16 + lr;
      float bv = bias ? bias[n] : 0.0f;
#pragma unroll
      for (int v = 0; v < 8; v++) {
        int m = m0 + ms * 16 + v + 8 * hi;
        float val = acc[ms][ns][v] + bv;
        size_t o = (size_t)m * N + n;
        if (resid) val += resid[o];
        if (Cf) Cf[o] = val;
        if (Ch) Ch[o] = (_Float16)val;
      }
    }
  }
}

// ---------------------------------------------------------------------------
// Depthwise causal conv (k=4) + bias + SiLU. xp = first 512 cols of xz.
// ---------------------------------------------------------------------------
__global__ void k_conv_silu(const float* __restrict__ xz,
                            const float* __restrict__ conv_w,
                            const float* __restrict__ conv_b,
                            float* __restrict__ u,
                            _Float16* __restrict__ u16) {
  int idx = blockIdx.x * blockDim.x + threadIdx.x;
  if (idx >= MROWS * D_INNER) return;
  int d   = idx & (D_INNER - 1);
  int row = idx >> 9;
  int t   = row & (SEQL - 1);
  float acc = conv_b[d];
#pragma unroll
  for (int j = 0; j < D_CONVK; j++) {
    int tj = t - (D_CONVK - 1) + j;
    if (tj >= 0)
      acc += xz[(size_t)(row - (D_CONVK - 1) + j) * (2 * D_INNER) + d] * conv_w[d * D_CONVK + j];
  }
  float sv = acc / (1.0f + __expf(-acc));
  u[idx]   = sv;
  u16[idx] = (_Float16)sv;
}

// ---------------------------------------------------------------------------
// dt_proj (K=16) + bias + softplus -> delta. One block per row.
// ---------------------------------------------------------------------------
__global__ void k_dtproj_softplus(const float* __restrict__ dbl,
                                  const float* __restrict__ dtw,
                                  const float* __restrict__ dtb,
                                  float* __restrict__ delta) {
  __shared__ float sdt[DT_RANK];
  int m = blockIdx.x;
  if (threadIdx.x < DT_RANK) sdt[threadIdx.x] = dbl[(size_t)m * 48 + threadIdx.x];
  __syncthreads();
  for (int n = threadIdx.x; n < D_INNER; n += blockDim.x) {
    float acc = dtb[n];
#pragma unroll
    for (int r = 0; r < DT_RANK; r++) acc += sdt[r] * dtw[n * DT_RANK + r];
    float sp = (acc > 20.0f) ? acc : __logf(1.0f + __expf(acc));
    delta[(size_t)m * D_INNER + n] = sp;
  }
}

// ---------------------------------------------------------------------------
// Selective scan: one lane per (b, d, s); 16-lane shuffle reduction for h*C.
// Fuses +u*D and the SiLU(z) gate; writes y as f16 for out_proj.
// ---------------------------------------------------------------------------
__global__ void k_selective_scan(const float* __restrict__ delta,
                                 const float* __restrict__ u,
                                 const float* __restrict__ dbl,
                                 const float* __restrict__ xz,
                                 const float* __restrict__ A_log,
                                 const float* __restrict__ Dp,
                                 _Float16* __restrict__ y16) {
  int g = blockIdx.x * blockDim.x + threadIdx.x;       // [0, 8*512*16)
  if (g >= BATCH * D_INNER * D_STATE) return;
  int s = g & (D_STATE - 1);
  int d = (g >> 4) & (D_INNER - 1);
  int b = g >> 13;
  float A  = -__expf(A_log[d * D_STATE + s]);
  float Dd = Dp[d];
  float h = 0.0f;
  size_t rb = (size_t)b * SEQL;
  for (int t = 0; t < SEQL; t++) {
    size_t row = rb + t;
    float dl = delta[row * D_INNER + d];
    float uu = u[row * D_INNER + d];
    float Bt = dbl[row * 48 + 16 + s];
    float Ct = dbl[row * 48 + 32 + s];
    h = __expf(dl * A) * h + (dl * uu) * Bt;
    float p = h * Ct;
    p += __shfl_xor(p, 1, 32);
    p += __shfl_xor(p, 2, 32);
    p += __shfl_xor(p, 4, 32);
    p += __shfl_xor(p, 8, 32);
    if (s == 0) {
      float z = xz[row * (2 * D_INNER) + D_INNER + d];
      float y = (p + uu * Dd) * (z / (1.0f + __expf(-z)));
      y16[row * D_INNER + d] = (_Float16)y;
    }
  }
}

// ---------------------------------------------------------------------------
// SwiGLU gating: g = silu(uv[:, :1024]) * uv[:, 1024:] -> f16
// ---------------------------------------------------------------------------
__global__ void k_swiglu(const float* __restrict__ uv, _Float16* __restrict__ g) {
  int idx = blockIdx.x * blockDim.x + threadIdx.x;
  if (idx >= MROWS * MLP_H) return;
  int m = idx >> 10;
  int hh = idx & (MLP_H - 1);
  float a = uv[(size_t)m * (2 * MLP_H) + hh];
  float v = uv[(size_t)m * (2 * MLP_H) + MLP_H + hh];
  g[idx] = (_Float16)((a / (1.0f + __expf(-a))) * v);
}

// ---------------------------------------------------------------------------
// host launcher
// ---------------------------------------------------------------------------
extern "C" void kernel_launch(void* const* d_in, const int* in_sizes, int n_in,
                              void* d_out, int out_size, void* d_ws, size_t ws_size,
                              hipStream_t stream) {
  const float* x         = (const float*)d_in[0];
  const float* norm1_w   = (const float*)d_in[1];
  const float* in_proj_w = (const float*)d_in[2];   // (1024, 256)
  const float* conv_w    = (const float*)d_in[3];   // (512, 4)
  const float* conv_b    = (const float*)d_in[4];
  const float* x_proj_w  = (const float*)d_in[5];   // (48, 512)
  const float* dt_proj_w = (const float*)d_in[6];   // (512, 16)
  const float* dt_proj_b = (const float*)d_in[7];
  const float* A_log     = (const float*)d_in[8];   // (512, 16)
  const float* Dp        = (const float*)d_in[9];
  const float* out_proj_w= (const float*)d_in[10];  // (256, 512)
  const float* norm2_w   = (const float*)d_in[11];
  const float* fc1_w     = (const float*)d_in[12];  // (2048, 256)
  const float* fc1_b     = (const float*)d_in[13];
  const float* fc2_w     = (const float*)d_in[14];  // (256, 1024)
  const float* fc2_b     = (const float*)d_in[15];
  float* out = (float*)d_out;

  // bump allocator on d_ws (256B aligned)
  char* wsb = (char*)d_ws;
  size_t off = 0;
  auto alloc = [&](size_t bytes) -> void* {
    off = (off + 255) & ~(size_t)255;
    void* p = wsb + off;
    off += bytes;
    return p;
  };

  _Float16* w_inT  = (_Float16*)alloc((size_t)256  * 1024 * 2); // K x N
  _Float16* w_xpT  = (_Float16*)alloc((size_t)512  * 48   * 2);
  _Float16* w_opT  = (_Float16*)alloc((size_t)512  * 256  * 2);
  _Float16* w_f1T  = (_Float16*)alloc((size_t)256  * 2048 * 2);
  _Float16* w_f2T  = (_Float16*)alloc((size_t)1024 * 256  * 2);
  _Float16* xn16   = (_Float16*)alloc((size_t)MROWS * D_MODEL * 2);
  float*    xz     = (float*)   alloc((size_t)MROWS * 2 * D_INNER * 4);
  float*    u      = (float*)   alloc((size_t)MROWS * D_INNER * 4);
  _Float16* u16    = (_Float16*)alloc((size_t)MROWS * D_INNER * 2);
  float*    dbl    = (float*)   alloc((size_t)MROWS * 48 * 4);
  float*    delta  = (float*)   alloc((size_t)MROWS * D_INNER * 4);
  _Float16* y16    = (_Float16*)alloc((size_t)MROWS * D_INNER * 2);
  float*    x2     = (float*)   alloc((size_t)MROWS * D_MODEL * 4);
  _Float16* xn2_16 = (_Float16*)alloc((size_t)MROWS * D_MODEL * 2);
  float*    uv     = (float*)   alloc((size_t)MROWS * 2 * MLP_H * 4);
  _Float16* g16    = (_Float16*)alloc((size_t)MROWS * MLP_H * 2);
  (void)ws_size; (void)in_sizes; (void)n_in; (void)out_size;

  auto cdiv = [](long a, long b) { return (int)((a + b - 1) / b); };

  // 1) weight convert+transpose
  k_convert_transpose<<<cdiv(1024L * 256, 256), 256, 0, stream>>>(in_proj_w,  w_inT, 1024, 256);
  k_convert_transpose<<<cdiv(48L   * 512, 256), 256, 0, stream>>>(x_proj_w,   w_xpT, 48,   512);
  k_convert_transpose<<<cdiv(256L  * 512, 256), 256, 0, stream>>>(out_proj_w, w_opT, 256,  512);
  k_convert_transpose<<<cdiv(2048L * 256, 256), 256, 0, stream>>>(fc1_w,      w_f1T, 2048, 256);
  k_convert_transpose<<<cdiv(256L * 1024, 256), 256, 0, stream>>>(fc2_w,      w_f2T, 256,  1024);

  // 32x64 tiles for all N%64==0 GEMMs
  auto gemm64 = [&](const _Float16* A, const _Float16* Bt, const float* bias,
                    const float* resid, float* Cf, _Float16* Ch, int M, int N, int K) {
    long tiles = (long)(M / 32) * (N / 64);
    int blocks = cdiv(tiles, 8);                 // 8 waves per 256-thread block
    k_wmma_gemm<2, 4><<<blocks, 256, 0, stream>>>(A, Bt, bias, resid, Cf, Ch, M, N, K);
  };
  // 32x48 tiles for x_proj (N=48)
  auto gemm48 = [&](const _Float16* A, const _Float16* Bt, float* Cf, int M, int N, int K) {
    long tiles = (long)(M / 32) * (N / 48);
    int blocks = cdiv(tiles, 8);
    k_wmma_gemm<2, 3><<<blocks, 256, 0, stream>>>(A, Bt, nullptr, nullptr, Cf, nullptr, M, N, K);
  };

  // 2) rmsnorm1
  k_rmsnorm_f16<<<cdiv((long)MROWS, 8), 256, 0, stream>>>(x, norm1_w, xn16, MROWS);
  // 3) in_proj: xz = xn @ in_proj_w^T
  gemm64(xn16, w_inT, nullptr, nullptr, xz, nullptr, MROWS, 2 * D_INNER, D_MODEL);
  // 4) conv + silu -> u
  k_conv_silu<<<cdiv((long)MROWS * D_INNER, 256), 256, 0, stream>>>(xz, conv_w, conv_b, u, u16);
  // 5) x_proj: dbl = u @ x_proj_w^T   (N=48)
  gemm48(u16, w_xpT, dbl, MROWS, 48, D_INNER);
  // 6) dt_proj + softplus -> delta
  k_dtproj_softplus<<<MROWS, 256, 0, stream>>>(dbl, dt_proj_w, dt_proj_b, delta);
  // 7) selective scan (+ u*D, * silu(z)) -> y16
  k_selective_scan<<<cdiv((long)BATCH * D_INNER * D_STATE, 256), 256, 0, stream>>>(
      delta, u, dbl, xz, A_log, Dp, y16);
  // 8) out_proj + residual x -> x2
  gemm64(y16, w_opT, nullptr, x, x2, nullptr, MROWS, D_MODEL, D_INNER);
  // 9) rmsnorm2
  k_rmsnorm_f16<<<cdiv((long)MROWS, 8), 256, 0, stream>>>(x2, norm2_w, xn2_16, MROWS);
  // 10) fc1 + bias -> uv
  gemm64(xn2_16, w_f1T, fc1_b, nullptr, uv, nullptr, MROWS, 2 * MLP_H, D_MODEL);
  // 11) swiglu gate -> g16
  k_swiglu<<<cdiv((long)MROWS * MLP_H, 256), 256, 0, stream>>>(uv, g16);
  // 12) fc2 + bias + residual x2 -> out
  gemm64(g16, w_f2T, fc2_b, x2, out, nullptr, MROWS, D_MODEL, MLP_H);
}
